// AttentionModule_14336600834794
// MI455X (gfx1250) — matrix operand order repaired and checked by reading, not compiled, not run on hardware
//
#include <hip/hip_runtime.h>
#include <hip/hip_bf16.h>

typedef _Float16 v16h __attribute__((ext_vector_type(16)));
typedef _Float16 v8h  __attribute__((ext_vector_type(8)));
typedef _Float16 v4h  __attribute__((ext_vector_type(4)));
typedef float    v8f  __attribute__((ext_vector_type(8)));

#define B_SZ 128
#define D_SZ 128
#define Q_SZ 256
#define L_SZ 1024
#define TEMP 4.0f

__device__ inline float wave_max(float x) {
#pragma unroll
  for (int o = 16; o > 0; o >>= 1) x = fmaxf(x, __shfl_xor(x, o, 32));
  return x;
}
__device__ inline float wave_sum(float x) {
#pragma unroll
  for (int o = 16; o > 0; o >>= 1) x += __shfl_xor(x, o, 32);
  return x;
}
__device__ inline v16h cat8(v8h lo, v8h hi) {
  return __builtin_shufflevector(lo, hi, 0, 1, 2, 3, 4, 5, 6, 7,
                                 8, 9, 10, 11, 12, 13, 14, 15);
}

// ---------------------------------------------------------------------------
// Kernel 1: S = C^T Q (per batch, 32-row l-tile, full q=256), softmax over q,
//           write T*attn_q transposed into the map region as map[b][q][l].
// ---------------------------------------------------------------------------
__global__ __launch_bounds__(256) void k_scores(const float* __restrict__ qry,
                                                const float* __restrict__ ctx,
                                                float* __restrict__ map) {
  const int b    = blockIdx.y;
  const int l0   = blockIdx.x * 32;
  const int tid  = threadIdx.x;
  const int w    = tid >> 5;       // wave 0..7
  const int lane = tid & 31;
  const int half = lane >> 4;      // lane half 0/1
  const int l16  = lane & 15;

  __shared__ float    S[32][260];       // 33,280 B
  __shared__ _Float16 As[32][136];      //  8,704 B  [l][k], row stride 272 B
  __shared__ _Float16 Bs[256][40];      // 20,480 B  [q][k_chunk], row stride 80 B

  const float* ctxb = ctx + (size_t)b * D_SZ * L_SZ;
  const float* qryb = qry + (size_t)b * D_SZ * Q_SZ;

  // ---- Stage C tile (128 k x 32 l) -> As[l][k] f16.
  // Each thread: 4 float4 loads (4 consecutive k rows, same 4 l cols),
  // then 4 packed v4h (b64) transposed stores.
  {
    const int k4 = tid >> 3;            // 0..31 -> k rows k4*4 .. k4*4+3
    const int l4 = (tid & 7) * 4;       // 0..28
    float g[4][4];
#pragma unroll
    for (int i = 0; i < 4; ++i)
      *(float4*)g[i] =
          *(const float4*)(ctxb + (size_t)(k4 * 4 + i) * L_SZ + l0 + l4);
#pragma unroll
    for (int c = 0; c < 4; ++c) {
      v4h p;
#pragma unroll
      for (int i = 0; i < 4; ++i) p[i] = (_Float16)g[i][c];
      *(v4h*)(&As[l4 + c][k4 * 4]) = p;
    }
  }

  v8f acc[2][2] = {};                   // [m-tile][n-tile]

#pragma unroll
  for (int kc = 0; kc < 4; ++kc) {      // K = D = 128, chunks of 32
    __syncthreads();                    // As ready / previous Bs consumed
    // ---- Stage Q chunk (32 k x 256 q) -> Bs[q][k] f16.
    // 512 tasks (2/thread): task = (q4-group, kq); hoisted loads, b64 stores.
    float g[2][4][4];
#pragma unroll
    for (int j = 0; j < 2; ++j) {
      const int id = tid + 256 * j;
      const int kq = id & 7;            // k sub-block (4 rows)
      const int q4 = (id >> 3) * 4;     // 0..252
      const float* base = qryb + (size_t)(kc * 32 + kq * 4) * Q_SZ + q4;
#pragma unroll
      for (int i = 0; i < 4; ++i)
        *(float4*)g[j][i] = *(const float4*)(base + (size_t)i * Q_SZ);
    }
#pragma unroll
    for (int j = 0; j < 2; ++j) {
      const int id = tid + 256 * j;
      const int kq = id & 7;
      const int q4 = (id >> 3) * 4;
#pragma unroll
      for (int c = 0; c < 4; ++c) {
        v4h p;
#pragma unroll
        for (int i = 0; i < 4; ++i) p[i] = (_Float16)g[j][i][c];
        *(v4h*)(&Bs[q4 + c][kq * 4]) = p;
      }
    }
    __syncthreads();

    v16h a[2], bq[2];
#pragma unroll
    for (int mt = 0; mt < 2; ++mt) {    // A: elems 0..7 = k+0..7, 8..15 = k+16..23
      const _Float16* ap = &As[mt * 16 + l16][kc * 32 + half * 8];
      a[mt] = cat8(*(const v8h*)ap, *(const v8h*)(ap + 16));
    }
#pragma unroll
    for (int it = 0; it < 2; ++it) {    // B: elems 0..7 = k+0..7, 8..15 = k+16..23
      const _Float16* bp = &Bs[(w + it * 8) * 16 + l16][half * 8];
      bq[it] = cat8(*(const v8h*)bp, *(const v8h*)(bp + 16));
    }
#pragma unroll
    for (int mt = 0; mt < 2; ++mt)
#pragma unroll
      for (int it = 0; it < 2; ++it)
        acc[mt][it] = __builtin_amdgcn_wmma_f32_16x16x32_f16(
            false, a[mt], false, bq[it], (short)0, acc[mt][it], false, false);
  }

  // Spill accumulators to LDS: S[l_local][q]
#pragma unroll
  for (int mt = 0; mt < 2; ++mt)
#pragma unroll
    for (int it = 0; it < 2; ++it) {
      const int q = (w + it * 8) * 16 + l16;
#pragma unroll
      for (int r = 0; r < 8; ++r)
        S[mt * 16 + r + half * 8][q] = acc[mt][it][r];
    }
  __syncthreads();

  // Row softmax over q (256 wide). 32 rows / 8 waves = 4 rows per wave.
#pragma unroll
  for (int rr = 0; rr < 4; ++rr) {
    const int row = w * 4 + rr;
    float v[8];
    float m = -1e30f;
#pragma unroll
    for (int j = 0; j < 8; ++j) { v[j] = S[row][lane + 32 * j]; m = fmaxf(m, v[j]); }
    m = wave_max(m);
    float s = 0.f;
#pragma unroll
    for (int j = 0; j < 8; ++j) { v[j] = __expf(v[j] - m); s += v[j]; }
    s = wave_sum(s);
    const float inv = TEMP / s;         // pre-scale by temperature for kernel 2
#pragma unroll
    for (int j = 0; j < 8; ++j) S[row][lane + 32 * j] = v[j] * inv;
  }
  __syncthreads();

  // Transposed, coalesced write-out: lanes cover 32 consecutive l per q.
  for (int q = w; q < Q_SZ; q += 8)
    map[((size_t)b * Q_SZ + q) * L_SZ + l0 + lane] = S[lane][q];
}

// ---------------------------------------------------------------------------
// Kernel 2: in-place softmax over L=1024 for each (b,q) row of the map.
// ---------------------------------------------------------------------------
__global__ __launch_bounds__(256) void k_softmax_l(float* __restrict__ map) {
  float* p = map + (size_t)blockIdx.x * L_SZ;
  const int t    = threadIdx.x;
  const int w    = t >> 5;
  const int lane = t & 31;
  __shared__ float red[8];

  float4 v = ((const float4*)p)[t];
  float m = fmaxf(fmaxf(v.x, v.y), fmaxf(v.z, v.w));
  m = wave_max(m);
  if (lane == 0) red[w] = m;
  __syncthreads();
  float gm = red[0];
#pragma unroll
  for (int i = 1; i < 8; ++i) gm = fmaxf(gm, red[i]);
  __syncthreads();

  v.x = __expf(v.x - gm); v.y = __expf(v.y - gm);
  v.z = __expf(v.z - gm); v.w = __expf(v.w - gm);
  float s = wave_sum(v.x + v.y + v.z + v.w);
  if (lane == 0) red[w] = s;
  __syncthreads();
  float gs = 0.f;
#pragma unroll
  for (int i = 0; i < 8; ++i) gs += red[i];

  const float inv = 1.f / gs;
  v.x *= inv; v.y *= inv; v.z *= inv; v.w *= inv;
  ((float4*)p)[t] = v;
}

// ---------------------------------------------------------------------------
// Kernel 3: W = C * attn_c^T  (M=d, N=q, K=l=1024). Fragment data is
// contiguous in memory for both operands -> explicit float4 loads.
// ---------------------------------------------------------------------------
__global__ __launch_bounds__(256) void k_weighted(const float* __restrict__ ctx,
                                                  const float* __restrict__ map,
                                                  float* __restrict__ outw) {
  const int b    = blockIdx.y;
  const int d0   = blockIdx.x * 64;
  const int tid  = threadIdx.x;
  const int w    = tid >> 5;
  const int lane = tid & 31;
  const int half = lane >> 4;
  const int l16  = lane & 15;

  const float* ctxb = ctx + (size_t)b * D_SZ * L_SZ;
  const float* mapb = map + (size_t)b * Q_SZ * L_SZ;

  v8f acc[4][2] = {};                   // [m-tile][n-tile]

  for (int kc = 0; kc < 32; ++kc) {     // K = L = 1024, chunks of 32
    v16h a[4], bq[2];
    // A = C: per lane two 8-float runs at +0 and +16 (32B aligned).
#pragma unroll
    for (int mt = 0; mt < 4; ++mt) {
      const float* rowp =
          ctxb + (size_t)(d0 + mt * 16 + l16) * L_SZ + kc * 32 + half * 8;
      float g[4][4];
      *(float4*)g[0] = *(const float4*)(rowp + 0);
      *(float4*)g[1] = *(const float4*)(rowp + 4);
      *(float4*)g[2] = *(const float4*)(rowp + 16);
      *(float4*)g[3] = *(const float4*)(rowp + 20);
#pragma unroll
      for (int i = 0; i < 4; ++i)
#pragma unroll
        for (int c = 0; c < 4; ++c)
          a[mt][i * 4 + c] = (_Float16)g[i][c];
    }
    // B = attn_c^T: per lane two 8-float runs at +0 and +16 (32B aligned).
#pragma unroll
    for (int it = 0; it < 2; ++it) {
      const float* rowp = mapb + (size_t)((w + it * 8) * 16 + l16) * L_SZ +
                          kc * 32 + half * 8;
      float g[4][4];
#pragma unroll
      for (int i = 0; i < 4; ++i)
        *(float4*)g[i] = *(const float4*)(rowp + (i >> 1) * 16 + (i & 1) * 4);
#pragma unroll
      for (int i = 0; i < 4; ++i)
#pragma unroll
        for (int c = 0; c < 4; ++c)
          bq[it][i * 4 + c] = (_Float16)g[i][c];
    }
#pragma unroll
    for (int mt = 0; mt < 4; ++mt)
#pragma unroll
      for (int it = 0; it < 2; ++it)
        acc[mt][it] = __builtin_amdgcn_wmma_f32_16x16x32_f16(
            false, a[mt], false, bq[it], (short)0, acc[mt][it], false, false);
  }

  // Coalesced f32 store of W[b][d][q].
#pragma unroll
  for (int mt = 0; mt < 4; ++mt)
#pragma unroll
    for (int it = 0; it < 2; ++it) {
      const int q = (w + it * 8) * 16 + l16;
#pragma unroll
      for (int r = 0; r < 8; ++r) {
        const int d = d0 + mt * 16 + r + half * 8;
        outw[((size_t)b * D_SZ + d) * Q_SZ + q] = acc[mt][it][r];
      }
    }
}

// ---------------------------------------------------------------------------
extern "C" void kernel_launch(void* const* d_in, const int* in_sizes, int n_in,
                              void* d_out, int out_size, void* d_ws, size_t ws_size,
                              hipStream_t stream) {
  const float* qry = (const float*)d_in[0];   // (B, D, Q) fp32
  const float* ctx = (const float*)d_in[1];   // (B, D, H, W) fp32, L = H*W
  float* outw = (float*)d_out;                                  // (B, D, Q)
  float* map  = outw + (size_t)B_SZ * D_SZ * Q_SZ;              // (B, Q, L)

  k_scores   <<<dim3(L_SZ / 32, B_SZ), 256, 0, stream>>>(qry, ctx, map);
  k_softmax_l<<<B_SZ * Q_SZ,           256, 0, stream>>>(map);
  k_weighted <<<dim3(2, B_SZ),         256, 0, stream>>>(ctx, map, outw);
}
